// ContrastiveSWM_60138132078979
// MI455X (gfx1250) — matrix operand order repaired and checked by reading, not compiled, not run on hardware
//
#include <hip/hip_runtime.h>
#include <hip/hip_bf16.h>

typedef __bf16 bf16;
typedef __attribute__((ext_vector_type(16))) __bf16 v16bf;
typedef __attribute__((ext_vector_type(8)))  float  v8f;
typedef int v4i __attribute__((vector_size(16)));

#define LN_EPS 1e-5f

// element strides (padded to avoid LDS bank conflicts on b128 accesses)
#define ACT_STRIDE 264   // 256 activation cols, bf16
#define ST_STRIDE  136   // 128 state cols, bf16
#define NIN_STRIDE 424   // 416 node-input cols, bf16
#define W_STRIDE   40    // 32 staged K per chunk, bf16 (Bt layout: [n][k])
#define TMP_STRIDE 264   // f32 scratch rows

#define WBUF_ELEMS (256 * W_STRIDE)   // one weight-chunk buffer (bf16 elems)
#define NTILE_ELEMS (16 * W_STRIDE)   // one N-tile inside a chunk buffer

// LDS carve (bytes) — double-buffered weight chunks (contiguous)
#define OFF_ACT   0
#define OFF_W0    (OFF_ACT + 256 * ACT_STRIDE * 2)          // 135168
#define OFF_ST    (OFF_W0  + 2 * WBUF_ELEMS * 2)            // +40960
#define OFF_NIN   (OFF_ST  + 16  * ST_STRIDE * 2)           // +4352
#define OFF_TMP   (OFF_NIN + 16  * NIN_STRIDE * 2)          // +13568
#define OFF_P     (OFF_TMP + 16  * TMP_STRIDE * 4)          // +16896
#define SMEM_BYTES (OFF_P + (9 * 256 + 128) * 4)            // total 220672

// param-block offsets inside sP (floats)
#define PB_EB1 0
#define PB_EB2 256
#define PB_EG  512
#define PB_EBT 768
#define PB_EB3 1024
#define PB_NB1 1280
#define PB_NB2 1536
#define PB_NG  1792
#define PB_NBT 2048
#define PB_NB3 2304

// workspace carve (bf16 elements): repacked weight chunks, kc-major, [n][32] per chunk
#define RW_E1 0
#define RW_E2 65536
#define RW_E3 131072
#define RW_N2 196608
#define RW_N1 262144          // 13 chunks * 8192
#define RW_N3 368640          // 8 chunks * 4096

#if defined(__gfx1250__) && __has_builtin(__builtin_amdgcn_global_load_async_to_lds_b128)
#define USE_ASYNC 1
#else
#define USE_ASYNC 0
#endif

union Frag { v16bf v; uint4 q[2]; };

// A fragment (16x32 bf16) from a row-major bf16 LDS tile.
// Lanes L: M=L&15; lanes<16 hold K {0..7,16..23}, lanes>=16 hold K {8..15,24..31}.
__device__ inline v16bf ldsA(const bf16* rowBase, int stride, int kOff) {
  const int lane = threadIdx.x & 31;
  const int half = lane >> 4;
  const int m    = lane & 15;
  const bf16* p = rowBase + m * stride + kOff + half * 8;
  Frag f;
  f.q[0] = *(const uint4*)(p);
  f.q[1] = *(const uint4*)(p + 16);
  return f.v;
}

// B fragment (32x16 bf16) from N-major staged weights Bt[n][k], k contiguous.
__device__ inline v16bf ldsB(const bf16* bt) {
  const int lane = threadIdx.x & 31;
  const int half = lane >> 4;
  const int n    = lane & 15;
  const bf16* p = bt + n * W_STRIDE + half * 8;
  Frag f;
  f.q[0] = *(const uint4*)(p);
  f.q[1] = *(const uint4*)(p + 16);
  return f.v;
}

// Edge-layer-1 A fragment built on the fly from node states.
__device__ inline v16bf ldsA_edge(const bf16* sSt, int w, int kc) {
  const int lane = threadIdx.x & 31;
  const int half = lane >> 4;
  const int s    = lane & 15;
  int srcRow, colBase;
  if (kc < 4) { srcRow = w; colBase = kc * 32; }
  else {
    int j = s + (s >= w ? 1 : 0);
    if (s == 15) j = w;               // dummy slot: masked at aggregation
    srcRow = j; colBase = (kc - 4) * 32;
  }
  const bf16* p = sSt + srcRow * ST_STRIDE + colBase + half * 8;
  Frag f;
  f.q[0] = *(const uint4*)(p);
  f.q[1] = *(const uint4*)(p + 16);
  return f.v;
}

// Issue a global->LDS copy of one pre-packed chunk ([nRows][32] bf16 contiguous).
__device__ inline void stage_issue(const bf16* __restrict__ g, bf16* sWbuf, int nRows, int tid) {
  for (int f = tid; f < nRows * 4; f += 512) {
    const int n = f >> 2, seg = (f & 3) * 8;      // 16-byte segments
#if USE_ASYNC
    __builtin_amdgcn_global_load_async_to_lds_b128(
        (v4i*)(g + n * 32 + seg),
        (v4i*)(sWbuf + n * W_STRIDE + seg), 0, 0);
#else
    *(uint4*)(sWbuf + n * W_STRIDE + seg) = *(const uint4*)(g + n * 32 + seg);
#endif
  }
}

__device__ inline void async_join() {
#if USE_ASYNC
#if __has_builtin(__builtin_amdgcn_s_wait_asynccnt)
  __builtin_amdgcn_s_wait_asynccnt(0);
#else
  asm volatile("s_wait_asynccnt 0" ::: "memory");
#endif
#endif
  __syncthreads();
}

// Scheduling directive: pipeline the WMMA stream against DS reads.
// Masks: 0x008 = MFMA/WMMA group, 0x100 = DS-read group.
__device__ inline void sched_pipeline_hint() {
#if defined(__gfx1250__) && __has_builtin(__builtin_amdgcn_sched_group_barrier)
  __builtin_amdgcn_sched_group_barrier(0x100, 10, 0);   // A frag (2) + 4 B tiles (8)
#pragma unroll
  for (int i = 0; i < 12; ++i) {
    __builtin_amdgcn_sched_group_barrier(0x008, 1, 0);  // 1 WMMA
    __builtin_amdgcn_sched_group_barrier(0x100, 2, 0);  // next B tile (2 reads)
  }
  __builtin_amdgcn_sched_group_barrier(0x008, 4, 0);    // drain last 4 WMMAs
#endif
}

#define WMMA_BF16(A, B, C) \
  __builtin_amdgcn_wmma_f32_16x16x32_bf16(false, (A), false, (B), (short)0, (C), false, false)

#define V8F_ZERO ((v8f){0.f, 0.f, 0.f, 0.f, 0.f, 0.f, 0.f, 0.f})

// 256x256x256 edge layer: double-buffered async weight staging + 4-deep
// software pipeline on B fragments, enforced via sched_group_barrier.
template<typename AFn>
__device__ inline void gemm_edge_layer(const bf16* __restrict__ gW, bf16* sWbase,
                                       int tid, v8f* acc, AFn aload) {
  stage_issue(gW, sWbase, 256, tid);
  async_join();
  for (int kc = 0; kc < 8; ++kc) {
    if (kc + 1 < 8)
      stage_issue(gW + (size_t)(kc + 1) * 8192, sWbase + ((kc + 1) & 1) * WBUF_ELEMS, 256, tid);
    const v16bf a = aload(kc);
    const bf16* sWc = sWbase + (kc & 1) * WBUF_ELEMS;
    v16bf b0 = ldsB(sWc + 0 * NTILE_ELEMS);
    v16bf b1 = ldsB(sWc + 1 * NTILE_ELEMS);
    v16bf b2 = ldsB(sWc + 2 * NTILE_ELEMS);
    v16bf b3 = ldsB(sWc + 3 * NTILE_ELEMS);
#pragma unroll
    for (int nt = 0; nt < 16; ++nt) {
      acc[nt] = WMMA_BF16(a, b0, acc[nt]);
      b0 = b1; b1 = b2; b2 = b3;                       // renamed away by full unroll
      if (nt + 4 < 16) b3 = ldsB(sWc + (nt + 4) * NTILE_ELEMS);
    }
    sched_pipeline_hint();
    async_join();
  }
}

// 16-row node layer: wave w owns N-tile w; one WMMA per chunk.
template<int KCH, int NROWS>
__device__ inline v8f gemm_node_layer(const bf16* __restrict__ gW, bf16* sWbase,
                                      int tid, int w, const bf16* aBase, int aStride,
                                      bool active) {
  v8f acc = V8F_ZERO;
  stage_issue(gW, sWbase, NROWS, tid);
  async_join();
  for (int kc = 0; kc < KCH; ++kc) {
    if (kc + 1 < KCH)
      stage_issue(gW + (size_t)(kc + 1) * NROWS * 32, sWbase + ((kc + 1) & 1) * WBUF_ELEMS, NROWS, tid);
    if (active) {
      const v16bf a  = ldsA(aBase, aStride, kc * 32);
      const v16bf bb = ldsB(sWbase + (kc & 1) * WBUF_ELEMS + w * NTILE_ELEMS);
      acc = WMMA_BF16(a, bb, acc);
    }
    async_join();
  }
  return acc;
}

// ---------------- weight repack pre-kernels (f32 -> bf16 Bt chunks) ----------------
extern "C" __global__ void repack256(const float* __restrict__ W, bf16* __restrict__ dst) {
  const int idx = blockIdx.x * 256 + threadIdx.x;   // 65536 total
  const int kc = idx >> 13;
  const int n  = (idx >> 5) & 255;
  const int kk = idx & 31;
  dst[idx] = (bf16)W[(size_t)(kc * 32 + kk) * 256 + n];
}

extern "C" __global__ void repackN1(const float* __restrict__ W, bf16* __restrict__ dst) {
  const int idx = blockIdx.x * 256 + threadIdx.x;   // 13*8192 = 106496 total
  const int kc = idx >> 13;
  const int n  = (idx >> 5) & 255;
  const int kk = idx & 31;
  const int kp = kc * 32 + kk;                      // padded K: state|onehot+pad|agg
  float v = 0.f;
  if (kp < 132)       v = W[(size_t)kp * 256 + n];
  else if (kp >= 160) v = W[(size_t)(kp - 28) * 256 + n];
  dst[idx] = (bf16)v;
}

extern "C" __global__ void repackN3(const float* __restrict__ W, bf16* __restrict__ dst) {
  const int idx = blockIdx.x * 256 + threadIdx.x;   // 8*4096 = 32768 total
  const int kc = idx >> 12;
  const int n  = (idx >> 5) & 127;
  const int kk = idx & 31;
  dst[idx] = (bf16)W[(size_t)(kc * 32 + kk) * 128 + n];
}

// ---------------- fused GNN kernel: one graph per workgroup ----------------
extern "C" __global__ __launch_bounds__(512)
void cswm_fused(const float* __restrict__ states, const int* __restrict__ action,
                const bf16* __restrict__ rE1, const bf16* __restrict__ rE2,
                const bf16* __restrict__ rE3, const bf16* __restrict__ rN1,
                const bf16* __restrict__ rN2, const bf16* __restrict__ rN3,
                const float* __restrict__ eb1, const float* __restrict__ eb2,
                const float* __restrict__ eg,  const float* __restrict__ ebt,
                const float* __restrict__ eb3, const float* __restrict__ nb1,
                const float* __restrict__ nb2, const float* __restrict__ ng,
                const float* __restrict__ nbt, const float* __restrict__ nb3,
                float* __restrict__ out) {
  extern __shared__ char smem[];
  bf16*  sAct = (bf16*)(smem + OFF_ACT);
  bf16*  sW   = (bf16*)(smem + OFF_W0);     // 2 contiguous chunk buffers
  bf16*  sSt  = (bf16*)(smem + OFF_ST);
  bf16*  sNIn = (bf16*)(smem + OFF_NIN);
  float* sTmp = (float*)(smem + OFF_TMP);
  float* sP   = (float*)(smem + OFF_P);

  const int b    = blockIdx.x;
  const int tid  = threadIdx.x;
  const int w    = tid >> 5;       // wave id == M-tile == receiver node
  const int lane = tid & 31;
  const int half = lane >> 4;
  const int l15  = lane & 15;

  // ---- stage params + states (made visible by first barrier in layer pipeline) ----
  for (int i = tid; i < 256; i += 512) {
    sP[PB_EB1 + i] = eb1[i]; sP[PB_EB2 + i] = eb2[i];
    sP[PB_EG  + i] = eg[i];  sP[PB_EBT + i] = ebt[i];
    sP[PB_EB3 + i] = eb3[i]; sP[PB_NB1 + i] = nb1[i];
    sP[PB_NB2 + i] = nb2[i]; sP[PB_NG  + i] = ng[i];
    sP[PB_NBT + i] = nbt[i];
    if (i < 128) sP[PB_NB3 + i] = nb3[i];
  }
  const float* Sb = states + (size_t)b * 16 * 128;
  for (int f = tid; f < 2048; f += 512) {
    const int r = f >> 7, c = f & 127;
    sSt[r * ST_STRIDE + c] = (bf16)Sb[f];
  }

  v8f acc[16];
#pragma unroll
  for (int nt = 0; nt < 16; ++nt) acc[nt] = V8F_ZERO;

  // ================= edge layer 1 =================
  gemm_edge_layer(rE1, sW, tid, acc,
                  [&](int kc) { return ldsA_edge(sSt, w, kc); });
  { // relu(x + eb1) -> sAct (own rows)
    const int mrow = w * 16 + half * 8;
#pragma unroll
    for (int nt = 0; nt < 16; ++nt) {
      const int n = nt * 16 + l15;
      const float bias = sP[PB_EB1 + n];
#pragma unroll
      for (int v = 0; v < 8; ++v) {
        const float x = fmaxf(acc[nt][v] + bias, 0.f);
        sAct[(mrow + v) * ACT_STRIDE + n] = (bf16)x;
      }
      acc[nt] = V8F_ZERO;
    }
  }

  // ================= edge layer 2 (+ in-register LayerNorm + relu) =================
  gemm_edge_layer(rE2, sW, tid, acc,
                  [&](int kc) { return ldsA(sAct + w * 16 * ACT_STRIDE, ACT_STRIDE, kc * 32); });
  {
    float rs[8], rq[8];
#pragma unroll
    for (int v = 0; v < 8; ++v) { rs[v] = 0.f; rq[v] = 0.f; }
#pragma unroll
    for (int nt = 0; nt < 16; ++nt) {
      const int n = nt * 16 + l15;
      const float bias = sP[PB_EB2 + n];
#pragma unroll
      for (int v = 0; v < 8; ++v) {
        const float x = acc[nt][v] + bias;
        acc[nt][v] = x;
        rs[v] += x; rq[v] += x * x;
      }
    }
#pragma unroll
    for (int v = 0; v < 8; ++v) {
#pragma unroll
      for (int off = 1; off < 16; off <<= 1) {  // per-16-lane-half: rows differ per half
        rs[v] += __shfl_xor(rs[v], off);
        rq[v] += __shfl_xor(rq[v], off);
      }
    }
    float mean[8], inv[8];
#pragma unroll
    for (int v = 0; v < 8; ++v) {
      mean[v] = rs[v] * (1.f / 256.f);
      const float var = rq[v] * (1.f / 256.f) - mean[v] * mean[v];
      inv[v] = rsqrtf(var + LN_EPS);
    }
    const int mrow = w * 16 + half * 8;
#pragma unroll
    for (int nt = 0; nt < 16; ++nt) {
      const int n = nt * 16 + l15;
      const float g = sP[PB_EG + n], be = sP[PB_EBT + n];
#pragma unroll
      for (int v = 0; v < 8; ++v) {
        float y = (acc[nt][v] - mean[v]) * inv[v] * g + be;
        y = fmaxf(y, 0.f);
        sAct[(mrow + v) * ACT_STRIDE + n] = (bf16)y;
      }
      acc[nt] = V8F_ZERO;
    }
  }

  // ================= edge layer 3 + fused scatter-sum =================
  gemm_edge_layer(rE3, sW, tid, acc,
                  [&](int kc) { return ldsA(sAct + w * 16 * ACT_STRIDE, ACT_STRIDE, kc * 32); });
  { // column-sum of the 15 real edge rows (+15*bias) -> node input agg segment
#pragma unroll
    for (int nt = 0; nt < 16; ++nt) {
      const int n = nt * 16 + l15;
      float p = 0.f;
#pragma unroll
      for (int v = 0; v < 7; ++v) p += acc[nt][v];
      if (half == 0) p += acc[nt][7];     // M=15 (dummy) lives in half==1,v==7: exclude
      p += __shfl_xor(p, 16);
      if (half == 0) {
        const float a = p + 15.f * sP[PB_EB3 + n];
        sNIn[w * NIN_STRIDE + 160 + n] = (bf16)a;
      }
    }
  }
  { // node input cols 0..159: state | one-hot(action) padded to 32
    const int a = action[b];
    for (int f = tid; f < 2048; f += 512) {
      const int r = f >> 7, c = f & 127;
      sNIn[r * NIN_STRIDE + c] = sSt[r * ST_STRIDE + c];
    }
    for (int f = tid; f < 16 * 32; f += 512) {
      const int r = f >> 5, c = f & 31;
      sNIn[r * NIN_STRIDE + 128 + c] = (bf16)((c == a) ? 1.f : 0.f);
    }
  }

  // ================= node layer 1 =================
  v8f nacc = gemm_node_layer<13, 256>(rN1, sW, tid, w, sNIn, NIN_STRIDE, true);
  {
    const int n = w * 16 + l15;
    const float bias = sP[PB_NB1 + n];
#pragma unroll
    for (int v = 0; v < 8; ++v) {
      const float x = fmaxf(nacc[v] + bias, 0.f);
      sAct[(half * 8 + v) * ACT_STRIDE + n] = (bf16)x;
    }
  }

  // ================= node layer 2 (+ LN via LDS + relu) =================
  nacc = gemm_node_layer<8, 256>(rN2, sW, tid, w, sAct, ACT_STRIDE, true);
  {
    const int n = w * 16 + l15;
    const float bias = sP[PB_NB2 + n];
#pragma unroll
    for (int v = 0; v < 8; ++v)
      sTmp[(half * 8 + v) * TMP_STRIDE + n] = nacc[v] + bias;
  }
  __syncthreads();
  { // wave w normalizes node-row w
    float s = 0.f, q = 0.f;
    for (int c = lane; c < 256; c += 32) {
      const float x = sTmp[w * TMP_STRIDE + c];
      s += x; q += x * x;
    }
#pragma unroll
    for (int off = 1; off < 32; off <<= 1) { s += __shfl_xor(s, off); q += __shfl_xor(q, off); }
    const float mean = s * (1.f / 256.f);
    const float var  = q * (1.f / 256.f) - mean * mean;
    const float inv  = rsqrtf(var + LN_EPS);
    for (int c = lane; c < 256; c += 32) {
      const float x = sTmp[w * TMP_STRIDE + c];
      const float y = (x - mean) * inv * sP[PB_NG + c] + sP[PB_NBT + c];
      sAct[w * ACT_STRIDE + c] = (bf16)fmaxf(y, 0.f);
    }
  }

  // ================= node layer 3 -> out =================
  nacc = gemm_node_layer<8, 128>(rN3, sW, tid, w, sAct, ACT_STRIDE, (w < 8));
  if (w < 8) {
    const int n = w * 16 + l15;
    const float bias = sP[PB_NB3 + n];
    float* O = out + (size_t)b * 16 * 128;
#pragma unroll
    for (int v = 0; v < 8; ++v)
      O[(half * 8 + v) * 128 + n] = nacc[v] + bias;
  }
}

extern "C" void kernel_launch(void* const* d_in, const int* in_sizes, int n_in,
                              void* d_out, int out_size, void* d_ws, size_t ws_size,
                              hipStream_t stream) {
  (void)in_sizes; (void)n_in; (void)ws_size; (void)out_size;
  const float* states = (const float*)d_in[0];
  const int*   action = (const int*)d_in[1];
  const float* eW1 = (const float*)d_in[2];
  const float* eb1 = (const float*)d_in[3];
  const float* eW2 = (const float*)d_in[4];
  const float* eb2 = (const float*)d_in[5];
  const float* eg  = (const float*)d_in[6];
  const float* ebt = (const float*)d_in[7];
  const float* eW3 = (const float*)d_in[8];
  const float* eb3 = (const float*)d_in[9];
  const float* nW1 = (const float*)d_in[10];
  const float* nb1 = (const float*)d_in[11];
  const float* nW2 = (const float*)d_in[12];
  const float* nb2 = (const float*)d_in[13];
  const float* ng  = (const float*)d_in[14];
  const float* nbt = (const float*)d_in[15];
  const float* nW3 = (const float*)d_in[16];
  const float* nb3 = (const float*)d_in[17];
  float* out = (float*)d_out;

  bf16* ws  = (bf16*)d_ws;
  bf16* rE1 = ws + RW_E1;
  bf16* rE2 = ws + RW_E2;
  bf16* rE3 = ws + RW_E3;
  bf16* rN2 = ws + RW_N2;
  bf16* rN1 = ws + RW_N1;
  bf16* rN3 = ws + RW_N3;

  repack256<<<dim3(256), dim3(256), 0, stream>>>(eW1, rE1);
  repack256<<<dim3(256), dim3(256), 0, stream>>>(eW2, rE2);
  repack256<<<dim3(256), dim3(256), 0, stream>>>(eW3, rE3);
  repack256<<<dim3(256), dim3(256), 0, stream>>>(nW2, rN2);
  repackN1<<<dim3(416), dim3(256), 0, stream>>>(nW1, rN1);
  repackN3<<<dim3(128), dim3(256), 0, stream>>>(nW3, rN3);

  cswm_fused<<<dim3(1024), dim3(512), (size_t)SMEM_BYTES, stream>>>(
      states, action, rE1, rE2, rE3, rN1, rN2, rN3,
      eb1, eb2, eg, ebt, eb3, nb1, nb2, ng, nbt, nb3, out);
}
